// net_gcn_62362925138837
// MI455X (gfx1250) — matrix operand order repaired and checked by reading, not compiled, not run on hardware
//
#include <hip/hip_runtime.h>

typedef __attribute__((ext_vector_type(16))) _Float16 v16h;
typedef __attribute__((ext_vector_type(8)))  float    v8f;
typedef __attribute__((ext_vector_type(4)))  float    v4f;

#define DIM 128

// ---------------- degree histogram ----------------
__global__ void k_degrees(const int* __restrict__ src, const int* __restrict__ dst,
                          int* degOut, int* degIn, int E) {
  int e = blockIdx.x * blockDim.x + threadIdx.x;
  if (e < E) {
    atomicAdd(&degOut[src[e]], 1);
    atomicAdd(&degIn[dst[e]], 1);
  }
}

// ---------------- per-node scorer projections: a = h.We[0:128], b = h.We[128:256] ----
__global__ void k_proj(const float* __restrict__ h, const float* __restrict__ We,
                       float* aProj, float* bProj) {
  int n = blockIdx.x, t = threadIdx.x;          // 128 threads
  __shared__ float sa[DIM], sb[DIM];
  float x = h[(size_t)n * DIM + t];
  sa[t] = x * We[t];
  sb[t] = x * We[DIM + t];
  __syncthreads();
  for (int off = DIM / 2; off > 0; off >>= 1) {
    if (t < off) { sa[t] += sa[t + off]; sb[t] += sb[t + off]; }
    __syncthreads();
  }
  if (t == 0) { aProj[n] = sa[0]; bProj[n] = sb[0]; }
}

// ---------------- exclusive scan (single block, chunked Hillis-Steele) ----------------
__global__ void k_scan(const int* __restrict__ deg, int* rowPtr, int N) {
  __shared__ int buf[1024];
  __shared__ int carry;
  int t = threadIdx.x;
  if (t == 0) carry = 0;
  __syncthreads();
  for (int base = 0; base < N; base += 1024) {
    int v = (base + t < N) ? deg[base + t] : 0;
    buf[t] = v;
    __syncthreads();
    int acc = v;
    for (int off = 1; off < 1024; off <<= 1) {
      int add = (t >= off) ? buf[t - off] : 0;
      __syncthreads();
      acc += add;
      buf[t] = acc;
      __syncthreads();
    }
    int total = buf[1023];
    if (base + t < N) rowPtr[base + t] = carry + acc - v;   // exclusive
    __syncthreads();
    if (t == 0) carry += total;
    __syncthreads();
  }
  if (t == 0) rowPtr[N] = carry;
}

// ---------------- per-edge weight: sigmoid gate * mask * sym-norm ----------------
__global__ void k_edgew(const int* __restrict__ src, const int* __restrict__ dst,
                        const float* __restrict__ mask,
                        const float* __restrict__ aP, const float* __restrict__ bP,
                        const int* __restrict__ degOut, const int* __restrict__ degIn,
                        const float* __restrict__ We, const float* __restrict__ be,
                        float* edgeW, int E) {
  int e = blockIdx.x * blockDim.x + threadIdx.x;
  if (e >= E) return;
  int s = src[e], d = dst[e];
  float score = aP[s] + bP[d]
              + We[2 * DIM]     * log1pf((float)degOut[s])
              + We[2 * DIM + 1] * log1pf((float)degIn[d])
              + be[0];
  float sig = 1.f / (1.f + expf(-score));
  float m   = sig * mask[e];
  float is  = rsqrtf(fmaxf((float)degIn[s], 1.f));
  float id  = rsqrtf(fmaxf((float)degIn[d], 1.f));
  edgeW[e] = m * is * id;
}

// ---------------- CSR fill (by destination) ----------------
__global__ void k_fill(const int* __restrict__ src, const int* __restrict__ dst,
                       const float* __restrict__ edgeW, const int* __restrict__ rowPtr,
                       int* cursor, int* csrCol, float* csrVal, int E) {
  int e = blockIdx.x * blockDim.x + threadIdx.x;
  if (e >= E) return;
  int d = dst[e];
  int pos = rowPtr[d] + atomicAdd(&cursor[d], 1);
  csrCol[pos] = src[e];
  csrVal[pos] = edgeW[e];
}

// ---------------- weight prep: f32 [K][N] -> f16 transposed [Npad][K], zero-padded ----
__global__ void k_prepw(const float* __restrict__ W, _Float16* Wt, int Ncols, int Npad) {
  int idx = blockIdx.x * blockDim.x + threadIdx.x;
  if (idx >= Npad * DIM) return;
  int n = idx >> 7, k = idx & (DIM - 1);
  float v = (n < Ncols) ? W[(size_t)k * Ncols + n] : 0.f;
  Wt[(size_t)n * DIM + k] = (_Float16)v;
}

// ---------------- CSR aggregation: agg[n] = sum_e w[e] * x[src[e]] (no atomics) ------
__global__ void k_agg(const int* __restrict__ rowPtr, const int* __restrict__ csrCol,
                      const float* __restrict__ csrVal, const float* __restrict__ x,
                      float* __restrict__ agg) {
  int n = blockIdx.x, t = threadIdx.x;          // 128 threads = one feature row
  int i0 = rowPtr[n], i1 = rowPtr[n + 1];
  float acc = 0.f;
  for (int i = i0; i < i1; i++) {
    int s   = csrCol[i];
    float w = csrVal[i];
    acc = fmaf(w, x[(size_t)s * DIM + t], acc);
  }
  agg[(size_t)n * DIM + t] = acc;
}

// ---------------- WMMA node-apply: Out = relu(A @ Wt^T + bias) -----------------------
// One wave owns a 16-row block and all NT column tiles; K = 128 in 4 chunks of 32.
template <int NT>
__global__ void k_mm(const float* __restrict__ A, const _Float16* __restrict__ Wt,
                     const float* __restrict__ bias, float* __restrict__ Out,
                     int M, int outCols, int reluFlag) {
  int wave    = (blockIdx.x * blockDim.x + threadIdx.x) >> 5;
  int lane    = threadIdx.x & 31;
  int rowTile = wave;
  if (rowTile * 16 >= M) return;                 // whole-wave uniform: EXEC stays full

  int mi = lane & 15;                            // row within A tile / column within B tile
  int hi = lane >> 4;                            // lane half selects K sub-group
  int rowA = rowTile * 16 + mi;

  v8f acc[NT];
#pragma unroll
  for (int i = 0; i < NT; i++)
#pragma unroll
    for (int r = 0; r < 8; r++) acc[i][r] = 0.f;

#pragma unroll
  for (int k0 = 0; k0 < DIM; k0 += 32) {
    // A fragment (f32 -> f16 on the fly), contiguous K order, same as B:
    // lane half 0: K = k0+{0..15}; lane half 1: K = k0+{16..31}
    const float* ap = A + (size_t)rowA * DIM + k0 + hi * 16;
    v4f f0 = *(const v4f*)(ap);
    v4f f1 = *(const v4f*)(ap + 4);
    v4f f2 = *(const v4f*)(ap + 8);
    v4f f3 = *(const v4f*)(ap + 12);
    v16h a;
#pragma unroll
    for (int i = 0; i < 4; i++) {
      a[i]      = (_Float16)f0[i];
      a[4 + i]  = (_Float16)f1[i];
      a[8 + i]  = (_Float16)f2[i];
      a[12 + i] = (_Float16)f3[i];
    }
#pragma unroll
    for (int nt = 0; nt < NT; nt++) {
      // B fragment: column = mi, lanes 0-15 hold K=k0+0..15, lanes 16-31 K=k0+16..31.
      // Wt stored [N][K] f16 -> one aligned contiguous 32B (v16h) load.
      v16h b = *(const v16h*)(Wt + (size_t)(nt * 16 + mi) * DIM + k0 + hi * 16);
      acc[nt] = __builtin_amdgcn_wmma_f32_16x16x32_f16(
          false, a, false, b, (short)0, acc[nt], false, false);
    }
  }

  // D layout: VGPR r -> M = r + 8*hi ; N = lane&15
  int rbase = rowTile * 16 + hi * 8;
#pragma unroll
  for (int nt = 0; nt < NT; nt++) {
    int col = nt * 16 + mi;
    if (col >= outCols) continue;                // layer-3 pad guard (40 -> 48)
    float bv = bias[col];
#pragma unroll
    for (int r = 0; r < 8; r++) {
      int rr = rbase + r;
      if (rr >= M) continue;
      float v = acc[nt][r] + bv;
      if (reluFlag) v = fmaxf(v, 0.f);
      Out[(size_t)rr * outCols + col] = v;
    }
  }
}

extern "C" void kernel_launch(void* const* d_in, const int* in_sizes, int n_in,
                              void* d_out, int out_size, void* d_ws, size_t ws_size,
                              hipStream_t stream) {
  (void)n_in; (void)out_size; (void)ws_size;
  const float* h   = (const float*)d_in[0];
  const float* em  = (const float*)d_in[1];
  const int*   src = (const int*)d_in[2];
  const int*   dst = (const int*)d_in[3];
  const float* We  = (const float*)d_in[4];
  const float* be  = (const float*)d_in[5];
  const float* W0  = (const float*)d_in[6];
  const float* b0  = (const float*)d_in[7];
  const float* W1  = (const float*)d_in[8];
  const float* b1  = (const float*)d_in[9];
  const float* W2  = (const float*)d_in[10];
  const float* b2  = (const float*)d_in[11];
  float* out = (float*)d_out;

  const int N = in_sizes[0] / DIM;   // 50000
  const int E = in_sizes[2];         // 640000
  const int NCLS = 40, NPAD = 48;

  // workspace carve (256B aligned), ~60.2 MB total
  char* p = (char*)d_ws;
  auto carve = [&](size_t bytes) -> void* {
    void* r = (void*)p;
    p += (bytes + 255) & ~(size_t)255;
    return r;
  };
  float*    bufA   = (float*)carve((size_t)N * DIM * 4);
  float*    bufB   = (float*)carve((size_t)N * DIM * 4);
  int*      degO   = (int*)  carve((size_t)N * 4);
  int*      degI   = (int*)  carve((size_t)N * 4);
  float*    aP     = (float*)carve((size_t)N * 4);
  float*    bP     = (float*)carve((size_t)N * 4);
  int*      rowPtr = (int*)  carve((size_t)(N + 1) * 4);
  int*      cursor = (int*)  carve((size_t)N * 4);
  float*    edgeW  = (float*)carve((size_t)E * 4);
  int*      csrCol = (int*)  carve((size_t)E * 4);
  float*    csrVal = (float*)carve((size_t)E * 4);
  _Float16* wt0    = (_Float16*)carve((size_t)DIM * DIM * 2);
  _Float16* wt1    = (_Float16*)carve((size_t)DIM * DIM * 2);
  _Float16* wt2    = (_Float16*)carve((size_t)NPAD * DIM * 2);

  hipMemsetAsync(degO,   0, (size_t)N * 4, stream);
  hipMemsetAsync(degI,   0, (size_t)N * 4, stream);
  hipMemsetAsync(cursor, 0, (size_t)N * 4, stream);

  const int tb = 256;
  k_degrees<<<(E + tb - 1) / tb, tb, 0, stream>>>(src, dst, degO, degI, E);
  k_proj<<<N, DIM, 0, stream>>>(h, We, aP, bP);
  k_scan<<<1, 1024, 0, stream>>>(degI, rowPtr, N);
  k_edgew<<<(E + tb - 1) / tb, tb, 0, stream>>>(src, dst, em, aP, bP, degO, degI,
                                                We, be, edgeW, E);
  k_fill<<<(E + tb - 1) / tb, tb, 0, stream>>>(src, dst, edgeW, rowPtr, cursor,
                                               csrCol, csrVal, E);
  k_prepw<<<(DIM * DIM + tb - 1) / tb, tb, 0, stream>>>(W0, wt0, DIM, DIM);
  k_prepw<<<(DIM * DIM + tb - 1) / tb, tb, 0, stream>>>(W1, wt1, DIM, DIM);
  k_prepw<<<(NPAD * DIM + tb - 1) / tb, tb, 0, stream>>>(W2, wt2, NCLS, NPAD);

  int rowTiles = (N + 15) / 16;            // 3125 (exact)
  int mmBlocks = (rowTiles + 3) / 4;       // 4 waves / block

  // layer 1: aggregate h -> bufB, apply W0 -> bufA (relu)
  k_agg<<<N, DIM, 0, stream>>>(rowPtr, csrCol, csrVal, h, bufB);
  k_mm<8><<<mmBlocks, 128, 0, stream>>>(bufB, wt0, b0, bufA, N, DIM, 1);
  // layer 2
  k_agg<<<N, DIM, 0, stream>>>(rowPtr, csrCol, csrVal, bufA, bufB);
  k_mm<8><<<mmBlocks, 128, 0, stream>>>(bufB, wt1, b1, bufA, N, DIM, 1);
  // layer 3: 40 classes (padded to 48 cols, store-guarded), no relu
  k_agg<<<N, DIM, 0, stream>>>(rowPtr, csrCol, csrVal, bufA, bufB);
  k_mm<3><<<mmBlocks, 128, 0, stream>>>(bufB, wt2, b2, out, N, NCLS, 0);
}
